// EnhancedHeteroGNN_24678882083362
// MI455X (gfx1250) — matrix operand order repaired and checked by reading, not compiled, not run on hardware
//
#include <hip/hip_runtime.h>
#include <hip/hip_bf16.h>
#include <math.h>

// ---------------------------------------------------------------------------
// EnhancedHeteroGNN for MI455X (gfx1250), wave32 + WMMA bf16.
// Dense N x 128 x {128,64} GEMMs via v_wmma_f32_16x16x32_bf16 with
// pre-packed bf16 weight fragments (contiguous b128 loads, no guards).
// All row counts are multiples of 16 by construction.
// ---------------------------------------------------------------------------

typedef __attribute__((ext_vector_type(16))) __bf16 v16bf;
typedef __attribute__((ext_vector_type(8)))  float  v8f;

union Afrag { v16bf v; __bf16 b[16]; };
union Bfrag { v16bf v; uint4  q[2];  };

__device__ __forceinline__ unsigned short f2bf(float f) {
    unsigned u = __float_as_uint(f);
    unsigned r = u + 0x7FFFu + ((u >> 16) & 1u);   // round-to-nearest-even
    return (unsigned short)(r >> 16);
}

// Load one A fragment (16x32 bf16 layout) for lane (h,r): two contiguous
// 8-float runs at k0+8h and k0+16+8h.
__device__ __forceinline__ Afrag load_afrag(const float* __restrict__ xr, int k0, int h8) {
    float4 x0 = *(const float4*)(xr + k0 + h8);
    float4 x1 = *(const float4*)(xr + k0 + h8 + 4);
    float4 x2 = *(const float4*)(xr + k0 + 16 + h8);
    float4 x3 = *(const float4*)(xr + k0 + 16 + h8 + 4);
    Afrag a;
    a.b[0]=(__bf16)x0.x; a.b[1]=(__bf16)x0.y; a.b[2]=(__bf16)x0.z; a.b[3]=(__bf16)x0.w;
    a.b[4]=(__bf16)x1.x; a.b[5]=(__bf16)x1.y; a.b[6]=(__bf16)x1.z; a.b[7]=(__bf16)x1.w;
    a.b[8]=(__bf16)x2.x; a.b[9]=(__bf16)x2.y; a.b[10]=(__bf16)x2.z; a.b[11]=(__bf16)x2.w;
    a.b[12]=(__bf16)x3.x; a.b[13]=(__bf16)x3.y; a.b[14]=(__bf16)x3.z; a.b[15]=(__bf16)x3.w;
    return a;
}

// ---- pack row-major f32 W[K,NCOL] into WMMA B-fragment order (bf16) -------
// PW layout: [ct][kstep][lane][j], j..15 contiguous per lane (32B = 2 b128).
__global__ void pack_w(const float* __restrict__ W, unsigned short* __restrict__ PW,
                       int K, int NCOL) {
    int idx = blockIdx.x * blockDim.x + threadIdx.x;
    if (idx >= K * NCOL) return;
    int j    = idx & 15;
    int lane = (idx >> 4) & 31;
    int rest = idx >> 9;
    int ksteps = K >> 5;
    int step = rest % ksteps;
    int ct   = rest / ksteps;
    int k    = step * 32 + (j & 7) + (lane >> 4) * 8 + (j >> 3) * 16;     // B 32x16 bf16 layout
    int col  = ct * 16 + (lane & 15);
    PW[idx] = f2bf(W[(size_t)k * NCOL + col]);
}

// ------------------------- generic WMMA linear -----------------------------
// Y[N,NCOL] = act( X[N,K] @ W[K,NCOL] + bias ); N % 16 == 0.
// block = (NCOL/16) waves; wave w -> 16-col tile; blockIdx.x -> 16-row tile.
__global__ void lin_wmma(const float* __restrict__ X, const unsigned short* __restrict__ PW,
                         const float* __restrict__ bias, float* __restrict__ Y,
                         int K, int NCOL, int act) {
    int wave = threadIdx.x >> 5, lane = threadIdx.x & 31;
    int h = lane >> 4, r = lane & 15;
    int h8 = h * 8;
    int row0 = blockIdx.x * 16;
    const float* xr = X + (size_t)(row0 + r) * K;
    const uint4* bp = (const uint4*)(PW + ((size_t)wave * (K >> 5)) * 512 + lane * 16);
    v8f acc = {};
    for (int k0 = 0; k0 < K; k0 += 32) {
        Afrag a = load_afrag(xr, k0, h8);
        Bfrag b;
        b.q[0] = bp[0]; b.q[1] = bp[1];
        bp += 64;                                    // next 512-bf16 step
        acc = __builtin_amdgcn_wmma_f32_16x16x32_bf16(false, a.v, false, b.v,
                                                      (short)0, acc, false, false);
    }
    int col = wave * 16 + r;
    float bv = bias ? bias[col] : 0.f;
#pragma unroll
    for (int i = 0; i < 8; ++i) {                    // C: row=i+8h, col=r
        float v = acc[i] + bv;
        if (act) v = fmaxf(v, 0.f);
        Y[(size_t)(row0 + i + h8) * NCOL + col] = v;
    }
}

// ---------------- fused CMA: q/k/v WMMA -> per-node 8x8 attention ----------
__global__ void cma_qkv_att(const float* __restrict__ X,
                            const unsigned short* __restrict__ PQ, const float* __restrict__ BQ,
                            const unsigned short* __restrict__ PK, const float* __restrict__ BK,
                            const unsigned short* __restrict__ PV, const float* __restrict__ BV,
                            float* __restrict__ ATT) {
    __shared__ float sm[3][16][128];   // q,k,v tiles
    __shared__ float sc[16][64];       // scores
    int wave = threadIdx.x >> 5, lane = threadIdx.x & 31;
    int h = lane >> 4, r = lane & 15;
    int h8 = h * 8;
    int row0 = blockIdx.x * 16;
    const float* xr = X + (size_t)(row0 + r) * 128;
    const unsigned short* Ps[3] = {PQ, PK, PV};
    const float* Bs[3] = {BQ, BK, BV};
    for (int m = 0; m < 3; ++m) {
        const uint4* bp = (const uint4*)(Ps[m] + ((size_t)wave * 4) * 512 + lane * 16);
        v8f acc = {};
        for (int k0 = 0; k0 < 128; k0 += 32) {
            Afrag a = load_afrag(xr, k0, h8);
            Bfrag b;
            b.q[0] = bp[0]; b.q[1] = bp[1];
            bp += 64;
            acc = __builtin_amdgcn_wmma_f32_16x16x32_bf16(false, a.v, false, b.v,
                                                          (short)0, acc, false, false);
        }
        int col = wave * 16 + r;
        float bv = Bs[m][col];
#pragma unroll
        for (int i = 0; i < 8; ++i) sm[m][i + h8][col] = acc[i] + bv;
    }
    __syncthreads();
    int t = threadIdx.x;
#pragma unroll
    for (int q = 0; q < 4; ++q) {                       // 1024 scores, 4/thread
        int idx = t * 4 + q;
        int node = idx >> 6, hg = idx & 63, hh = hg >> 3, g = hg & 7;
        float s = 0.f;
#pragma unroll
        for (int d = 0; d < 16; ++d) s += sm[0][node][hh * 16 + d] * sm[1][node][g * 16 + d];
        sc[node][hg] = s * 0.25f;                       // 1/sqrt(16)
    }
    __syncthreads();
    if (t < 128) {                                      // softmax: 16 nodes x 8 heads
        int node = t >> 3, hh = t & 7;
        float mx = -1e30f;
        for (int g = 0; g < 8; ++g) mx = fmaxf(mx, sc[node][hh * 8 + g]);
        float den = 0.f;
        for (int g = 0; g < 8; ++g) { float e = __expf(sc[node][hh * 8 + g] - mx); sc[node][hh * 8 + g] = e; den += e; }
        float inv = 1.f / den;
        for (int g = 0; g < 8; ++g) sc[node][hh * 8 + g] *= inv;
    }
    __syncthreads();
#pragma unroll
    for (int q = 0; q < 8; ++q) {                       // 2048 outputs, 8/thread
        int idx = t * 8 + q;
        int node = idx >> 7, c = idx & 127, hh = c >> 4, d = c & 15;
        float a = 0.f;
#pragma unroll
        for (int g = 0; g < 8; ++g) a += sc[node][hh * 8 + g] * sm[2][node][g * 16 + d];
        ATT[(size_t)(row0 + node) * 128 + c] = a;
    }
}

// ---------------- LayerNorm (+residual, +relu), 1 row / block --------------
__global__ void ln_residual(const float* __restrict__ A, const float* __restrict__ R,
                            const float* __restrict__ G, const float* __restrict__ Bb,
                            float* __restrict__ Y, int D, int relu) {
    __shared__ float red[128];
    int row = blockIdx.x, t = threadIdx.x;
    float z = A[(size_t)row * D + t] + (R ? R[(size_t)row * D + t] : 0.f);
    red[t] = z; __syncthreads();
    for (int s = D >> 1; s > 0; s >>= 1) { if (t < s) red[t] += red[t + s]; __syncthreads(); }
    float mean = red[0] / (float)D; __syncthreads();
    float dv = z - mean;
    red[t] = dv * dv; __syncthreads();
    for (int s = D >> 1; s > 0; s >>= 1) { if (t < s) red[t] += red[t + s]; __syncthreads(); }
    float var = red[0] / (float)D;
    float y = G[t] * dv * rsqrtf(var + 1e-5f) + Bb[t];
    if (relu) y = fmaxf(y, 0.f);
    Y[(size_t)row * D + t] = y;
}

// GNN layer epilogue: XS = relu(LN(new + bias5[type] + XS)) with per-type ln
__global__ void gnn_ln(const float* __restrict__ NEWB, const float* __restrict__ BIAS5,
                       const float* __restrict__ lnG, const float* __restrict__ lnB,
                       float* __restrict__ XS) {
    __shared__ float red[128];
    int row = blockIdx.x, t = threadIdx.x;
    int type = row < 200000 ? 0 : row < 220000 ? 1 : row < 222000 ? 2 : row < 234000 ? 3 : 4;
    float z = NEWB[(size_t)row * 128 + t] + BIAS5[type * 128 + t] + XS[(size_t)row * 128 + t];
    red[t] = z; __syncthreads();
    for (int s = 64; s > 0; s >>= 1) { if (t < s) red[t] += red[t + s]; __syncthreads(); }
    float mean = red[0] * (1.f / 128.f); __syncthreads();
    float dv = z - mean;
    red[t] = dv * dv; __syncthreads();
    for (int s = 64; s > 0; s >>= 1) { if (t < s) red[t] += red[t + s]; __syncthreads(); }
    float var = red[0] * (1.f / 128.f);
    float y = lnG[type * 128 + t] * dv * rsqrtf(var + 1e-5f) + lnB[type * 128 + t];
    XS[(size_t)row * 128 + t] = fmaxf(y, 0.f);
}

// ---------------- GAT helpers ----------------------------------------------
// Weff[k,h] = sum_d W[k, h*16+d] * a[h*16+d]   (collapses attention logit GEMM)
__global__ void build_weff(const float* __restrict__ W, const float* __restrict__ av,
                           float* __restrict__ Weff) {
    int k = threadIdx.x;
    if (k < 128) {
        for (int h = 0; h < 8; ++h) {
            float s = 0.f;
            for (int d = 0; d < 16; ++d) s += W[(size_t)k * 128 + h * 16 + d] * av[h * 16 + d];
            Weff[k * 8 + h] = s;
        }
    }
}

// AL[n,h] = X[n,:] @ Weff[:,h]
__global__ void node_al(const float* __restrict__ X, const float* __restrict__ Weff,
                        float* __restrict__ AL, int N) {
    int i = blockIdx.x * blockDim.x + threadIdx.x;
    if (i < N * 8) {
        int n = i >> 3, h = i & 7;
        float s = 0.f;
        const float4* xr = (const float4*)(X + (size_t)n * 128);
#pragma unroll 8
        for (int k4 = 0; k4 < 32; ++k4) {
            float4 x = xr[k4];
            s += x.x * Weff[(k4 * 4 + 0) * 8 + h] + x.y * Weff[(k4 * 4 + 1) * 8 + h]
               + x.z * Weff[(k4 * 4 + 2) * 8 + h] + x.w * Weff[(k4 * 4 + 3) * 8 + h];
        }
        AL[i] = s;
    }
}

__device__ __forceinline__ unsigned fenc(float f) {
    unsigned i = __float_as_uint(f);
    return (i & 0x80000000u) ? ~i : (i | 0x80000000u);
}
__device__ __forceinline__ float fdec(unsigned u) {
    if (u == 0u) return 0.f;   // untouched segment -> matches where(isfinite(m),m,0)
    return (u & 0x80000000u) ? __uint_as_float(u & 0x7FFFFFFFu) : __uint_as_float(~u);
}

__global__ void edge_logits_max(const int* __restrict__ src, const int* __restrict__ dst,
                                const float* __restrict__ ALS, const float* __restrict__ ALD,
                                float* __restrict__ E, unsigned* __restrict__ M, int nE) {
    long long i = (long long)blockIdx.x * blockDim.x + threadIdx.x;
    if (i >= (long long)nE * 8) return;
    int e = (int)(i >> 3), h = (int)(i & 7);
    float v = ALS[src[e] * 8 + h] + ALD[dst[e] * 8 + h];
    v = v > 0.f ? v : 0.2f * v;                 // leaky_relu(0.2)
    E[i] = v;
    atomicMax(&M[dst[e] * 8 + h], fenc(v));
}

__global__ void edge_exp_sum(const int* __restrict__ dst, float* __restrict__ E,
                             const unsigned* __restrict__ M, float* __restrict__ DEN, int nE) {
    long long i = (long long)blockIdx.x * blockDim.x + threadIdx.x;
    if (i >= (long long)nE * 8) return;
    int e = (int)(i >> 3), h = (int)(i & 7);
    int d = dst[e];
    float ex = __expf(E[i] - fdec(M[d * 8 + h]));
    E[i] = ex;
    atomicAdd(&DEN[d * 8 + h], ex);
}

// 32 lanes / edge, 4 channels / lane
__global__ void edge_scatter(const int* __restrict__ src, const int* __restrict__ dst,
                             const float* __restrict__ E, const float* __restrict__ DEN,
                             const float* __restrict__ HS, float* __restrict__ NEWD, int nE) {
    long long gid = (long long)blockIdx.x * blockDim.x + threadIdx.x;
    int e = (int)(gid >> 5), lane = (int)(gid & 31);
    if (e >= nE) return;
    int s = src[e], d = dst[e];
    int h = lane >> 2;                          // 4 channels share one head
    float alpha = E[(long long)e * 8 + h] / (DEN[d * 8 + h] + 1e-16f);
    int c0 = lane * 4;
    float4 hs = *(const float4*)(HS + (size_t)s * 128 + c0);
    float* dp = NEWD + (size_t)d * 128 + c0;
    atomicAdd(dp + 0, alpha * hs.x);
    atomicAdd(dp + 1, alpha * hs.y);
    atomicAdd(dp + 2, alpha * hs.z);
    atomicAdd(dp + 3, alpha * hs.w);
}

__global__ void bias_sums(const float* __restrict__ cb, float* __restrict__ BIAS5, int l) {
    int t = threadIdx.x;
    const int ndst[9] = {1, 0, 1, 3, 1, 2, 1, 4, 1};
    float acc[5] = {0, 0, 0, 0, 0};
    for (int e = 0; e < 9; ++e) acc[ndst[e]] += cb[(size_t)(l * 9 + e) * 128 + t];
    for (int ty = 0; ty < 5; ++ty) BIAS5[ty * 128 + t] = acc[ty];
}

// ---------------- prediction head tail -------------------------------------
__global__ void pred_final(const float* __restrict__ H, const float* __restrict__ w2,
                           const float* __restrict__ b2, const float* __restrict__ X,
                           const float* __restrict__ rw, const float* __restrict__ rb,
                           float* __restrict__ out, int B) {
    int row = blockIdx.x * 8 + (threadIdx.x >> 5);
    int lane = threadIdx.x & 31;
    if (row >= B) return;
    float s = 0.f;
    for (int c = lane; c < 64; c += 32)  s += H[(size_t)row * 64 + c] * w2[c];
    for (int c = lane; c < 128; c += 32) s += X[(size_t)row * 128 + c] * rw[c];
    for (int o = 16; o; o >>= 1) s += __shfl_down(s, o, 32);
    if (lane == 0) out[row] = fmaxf(s + b2[0] + rb[0], 0.f);
}

// ---------------------------------------------------------------------------
static inline void launch_lin(const float* X, const unsigned short* PW, const float* b,
                              float* Y, int N, int K, int NCOL, int act, hipStream_t s) {
    dim3 grid(N / 16), block((NCOL / 16) * 32);
    hipLaunchKernelGGL(lin_wmma, grid, block, 0, s, X, PW, b, Y, K, NCOL, act);
}
static inline void launch_pack(const float* W, unsigned short* PW, int K, int NCOL,
                               hipStream_t s) {
    hipLaunchKernelGGL(pack_w, dim3((K * NCOL + 255) / 256), dim3(256), 0, s, W, PW, K, NCOL);
}

extern "C" void kernel_launch(void* const* d_in, const int* in_sizes, int n_in,
                              void* d_out, int out_size, void* d_ws, size_t ws_size,
                              hipStream_t stream) {
    (void)n_in; (void)ws_size;
    const int NTt[5]    = {200000, 20000, 2000, 12000, 4000};
    const size_t off[6] = {0, 200000, 220000, 222000, 234000, 238000};
    const int Ntot = 238000;
    const int ET_si[9] = {0, 1, 1, 1, 3, 1, 2, 1, 4};
    const int ET_di[9] = {1, 0, 1, 3, 1, 2, 1, 4, 1};
    const int Kin[5]   = {64, 128, 64, 64, 64};

    // ---- inputs -----------------------------------------------------------
    const float* xin[5];
    for (int i = 0; i < 5; ++i) xin[i] = (const float*)d_in[i];
    const int* esrc[9]; const int* edst[9]; int nE[9];
    for (int t = 0; t < 9; ++t) {
        esrc[t] = (const int*)d_in[5 + 2 * t];
        edst[t] = (const int*)d_in[6 + 2 * t];
        nE[t]   = in_sizes[5 + 2 * t];
    }
    const int B = out_size;  // batch_size (8192)

    // ---- params (jax pytree flatten order: sorted dict keys) --------------
    auto F = [&](int i) { return (const float*)d_in[i]; };
    const float *cma_bk = F(24), *cma_bo = F(25), *cma_bq = F(26), *cma_bv = F(27);
    const float *cma_lnb = F(28), *cma_lng = F(29);
    const float *cma_wk = F(30), *cma_wo = F(31), *cma_wq = F(32), *cma_wv = F(33);
    const float *conv_ad = F(34), *conv_as = F(35), *conv_bias = F(36);
    const float *conv_wd = F(37), *conv_ws = F(38);
    const float *gene_b1 = F(39), *gene_b2 = F(40), *gene_w1 = F(41), *gene_w2 = F(42);
    const float *go_b1 = F(43), *go_b2 = F(44), *go_w1 = F(45), *go_w2 = F(46);
    const float *ln_b = F(47), *ln_g = F(48);
    const float *pr_b1 = F(49), *pr_b2 = F(50), *pr_lnb = F(51), *pr_lng = F(52);
    const float *pr_w1 = F(53), *pr_w2 = F(54);
    const float *res_b = F(55), *res_w = F(56);
    const float *snp_b1 = F(57), *snp_b2 = F(58), *snp_w1 = F(59), *snp_w2 = F(60);

    // ---- workspace layout -------------------------------------------------
    const size_t NBIG = 238016ull * 128ull;
    float* ws = (float*)d_ws;
    float* XS  = ws;                  // node features [238000,128]
    float* Bf1 = XS + NBIG;           // NEW accumulator / CMA ATT
    float* Bf2 = Bf1 + NBIG;          // HS / CMA output
    float* EE  = Bf2 + NBIG;          // edge scratch [1e6, 8]
    float* ALS = EE + 8000000;        // [200016,8]
    float* ALD = ALS + 1600128;
    unsigned* Menc = (unsigned*)(ALD + 1600128);
    float* DEN = (float*)(Menc + 1600128);
    float* WES = DEN + 1600128;       // [128,8]
    float* WED = WES + 1024;
    float* BIAS5 = WED + 1024;        // [5,128]
    float* PH  = BIAS5 + 640;         // [8192,64]
    // bf16 packed weight fragments
    unsigned short* PKb = (unsigned short*)(PH + 8192 * 64);
    unsigned short* pk_snp_w1  = PKb;            // 64x128
    unsigned short* pk_gene_w1 = PKb + 8192;     // 128x128
    unsigned short* pk_go_w1   = PKb + 24576;    // 64x128
    unsigned short* pk_snp_w2  = PKb + 32768;    // 128x128
    unsigned short* pk_gene_w2 = PKb + 49152;
    unsigned short* pk_go_w2   = PKb + 65536;
    unsigned short* pk_wq      = PKb + 81920;
    unsigned short* pk_wk      = PKb + 98304;
    unsigned short* pk_wv      = PKb + 114688;
    unsigned short* pk_wo      = PKb + 131072;
    unsigned short* pk_pr_w1   = PKb + 147456;   // 128x64
    unsigned short* pk_conv    = PKb + 155648;   // 18 x 128x128

    // ---- 0. pack all GEMM weights to bf16 fragment order ------------------
    launch_pack(snp_w1, pk_snp_w1, 64, 128, stream);
    launch_pack(gene_w1, pk_gene_w1, 128, 128, stream);
    launch_pack(go_w1, pk_go_w1, 64, 128, stream);
    launch_pack(snp_w2, pk_snp_w2, 128, 128, stream);
    launch_pack(gene_w2, pk_gene_w2, 128, 128, stream);
    launch_pack(go_w2, pk_go_w2, 128, 128, stream);
    launch_pack(cma_wq, pk_wq, 128, 128, stream);
    launch_pack(cma_wk, pk_wk, 128, 128, stream);
    launch_pack(cma_wv, pk_wv, 128, 128, stream);
    launch_pack(cma_wo, pk_wo, 128, 128, stream);
    launch_pack(pr_w1, pk_pr_w1, 128, 64, stream);
    for (int lt = 0; lt < 18; ++lt)
        launch_pack(conv_ws + (size_t)lt * 16384, pk_conv + (size_t)lt * 16384, 128, 128, stream);

    // ---- 1. per-type MLP encoders into XS ---------------------------------
    const unsigned short* pw1[5] = {pk_snp_w1, pk_gene_w1, pk_go_w1, pk_go_w1, pk_go_w1};
    const float* b1s[5] = {snp_b1, gene_b1, go_b1, go_b1, go_b1};
    const unsigned short* pw2[5] = {pk_snp_w2, pk_gene_w2, pk_go_w2, pk_go_w2, pk_go_w2};
    const float* b2s[5] = {snp_b2, gene_b2, go_b2, go_b2, go_b2};
    for (int i = 0; i < 5; ++i) {
        launch_lin(xin[i], pw1[i], b1s[i], Bf1, NTt[i], Kin[i], 128, 1, stream);
        launch_lin(Bf1, pw2[i], b2s[i], XS + off[i] * 128, NTt[i], 128, 128, 0, stream);
    }

    // ---- 2. cross-modal attention -----------------------------------------
    hipLaunchKernelGGL(cma_qkv_att, dim3(Ntot / 16), dim3(256), 0, stream,
                       XS, pk_wq, cma_bq, pk_wk, cma_bk, pk_wv, cma_bv, Bf1);
    launch_lin(Bf1, pk_wo, cma_bo, Bf2, Ntot, 128, 128, 0, stream);
    hipLaunchKernelGGL(ln_residual, dim3(Ntot), dim3(128), 0, stream,
                       Bf2, XS, cma_lng, cma_lnb, XS, 128, 0);

    // ---- 3. two HeteroConv GAT layers -------------------------------------
    for (int l = 0; l < 2; ++l) {
        hipMemsetAsync(Bf1, 0, (size_t)Ntot * 128 * sizeof(float), stream);
        hipLaunchKernelGGL(bias_sums, dim3(1), dim3(128), 0, stream, conv_bias, BIAS5, l);
        for (int t = 0; t < 9; ++t) {
            int si = ET_si[t], di = ET_di[t];
            int Ns = NTt[si], Nd = NTt[di];
            const float* Wst = conv_ws + (size_t)(l * 9 + t) * 16384;
            const float* Wdt = conv_wd + (size_t)(l * 9 + t) * 16384;
            const float* ast = conv_as + (size_t)(l * 9 + t) * 128;
            const float* adt = conv_ad + (size_t)(l * 9 + t) * 128;
            const float* Xs = XS + off[si] * 128;
            const float* Xd = XS + off[di] * 128;
            hipLaunchKernelGGL(build_weff, dim3(1), dim3(128), 0, stream, Wst, ast, WES);
            hipLaunchKernelGGL(build_weff, dim3(1), dim3(128), 0, stream, Wdt, adt, WED);
            hipLaunchKernelGGL(node_al, dim3((Ns * 8 + 255) / 256), dim3(256), 0, stream, Xs, WES, ALS, Ns);
            hipLaunchKernelGGL(node_al, dim3((Nd * 8 + 255) / 256), dim3(256), 0, stream, Xd, WED, ALD, Nd);
            launch_lin(Xs, pk_conv + (size_t)(l * 9 + t) * 16384, nullptr, Bf2, Ns, 128, 128, 0, stream);
            hipMemsetAsync(Menc, 0, (size_t)Nd * 8 * sizeof(unsigned), stream);
            hipMemsetAsync(DEN, 0, (size_t)Nd * 8 * sizeof(float), stream);
            long long w8 = (long long)nE[t] * 8;
            hipLaunchKernelGGL(edge_logits_max, dim3((unsigned)((w8 + 255) / 256)), dim3(256), 0, stream,
                               esrc[t], edst[t], ALS, ALD, EE, Menc, nE[t]);
            hipLaunchKernelGGL(edge_exp_sum, dim3((unsigned)((w8 + 255) / 256)), dim3(256), 0, stream,
                               edst[t], EE, Menc, DEN, nE[t]);
            long long w32 = (long long)nE[t] * 32;
            hipLaunchKernelGGL(edge_scatter, dim3((unsigned)((w32 + 255) / 256)), dim3(256), 0, stream,
                               esrc[t], edst[t], EE, DEN, Bf2, Bf1 + off[di] * 128, nE[t]);
        }
        hipLaunchKernelGGL(gnn_ln, dim3(Ntot), dim3(128), 0, stream,
                           Bf1, BIAS5, ln_g + (size_t)l * 640, ln_b + (size_t)l * 640, XS);
    }

    // ---- 4. prediction head on first B SNP rows ---------------------------
    launch_lin(XS, pk_pr_w1, pr_b1, PH, B, 128, 64, 0, stream);
    hipLaunchKernelGGL(ln_residual, dim3(B), dim3(64), 0, stream,
                       PH, (const float*)nullptr, pr_lng, pr_lnb, PH, 64, 1);
    hipLaunchKernelGGL(pred_final, dim3((B + 7) / 8), dim3(256), 0, stream,
                       PH, pr_w2, pr_b2, XS, res_w, res_b, (float*)d_out, B);
}
